// liGRUFrame_18872086298656
// MI455X (gfx1250) — matrix operand order repaired and checked by reading, not compiled, not run on hardware
//
#include <hip/hip_runtime.h>
#include <hip/hip_bf16.h>
#include <math.h>

typedef __bf16 bf16_t;
typedef __attribute__((ext_vector_type(16))) __bf16 v16bf;
typedef __attribute__((ext_vector_type(8)))  __bf16 v8bf;
typedef __attribute__((ext_vector_type(8)))  float  v8f;

#define T_STEPS 512
#define BATCH   32
#define DIM     512
#define HID     512
#define G3      1536            // 3*H
#define M_ROWS  (T_STEPS*BATCH) // 16384
#define SP      520             // padded LDS row stride (elements): 16B aligned, bank-rotating

static __device__ __forceinline__ v16bf cat16(v8bf lo, v8bf hi) {
  v16bf r;
#pragma unroll
  for (int i = 0; i < 8; ++i) { r[i] = lo[i]; r[i + 8] = hi[i]; }
  return r;
}

static __device__ __forceinline__ v8f wmma_bf16(v16bf a, v16bf b, v8f c) {
  // D = A(16x32 bf16) x B(32x16 bf16) + C(16x16 f32)
  return __builtin_amdgcn_wmma_f32_16x16x32_bf16(false, a, false, b, (short)0, c,
                                                 false, false);
}

static __device__ __forceinline__ float sigmoidf_(float x) {
  return 1.0f / (1.0f + __expf(-x));
}

// ---------------------------------------------------------------------------
// fp32 -> bf16 convert
// ---------------------------------------------------------------------------
__global__ void f32_to_bf16_kernel(const float* __restrict__ src,
                                   bf16_t* __restrict__ dst, int n) {
  int i = blockIdx.x * blockDim.x + threadIdx.x;
  int stride = gridDim.x * blockDim.x;
  for (; i < n; i += stride) dst[i] = (bf16_t)src[i];
}

// ---------------------------------------------------------------------------
// Pack W (G3 x K, row-major fp32) into WMMA B-fragment order, bf16.
// Tile (nt,kt): lane L holds column n = nt*16+(L&15), k values kt*32 + h*8 + {0..7}
// then kt*32 + 16 + h*8 + {0..7}, h = (L<16 ? 0 : 1)  -- matches ISA 16-bit B 32x16 layout.
// ---------------------------------------------------------------------------
__global__ void pack_w_kernel(const float* __restrict__ W,
                              bf16_t* __restrict__ dst, int K) {
  const int KT = K >> 5;
  const int ntiles = (G3 / 16) * KT;
  int gid = blockIdx.x * blockDim.x + threadIdx.x;
  int tile = gid >> 5;
  if (tile >= ntiles) return;
  const int lane = gid & 31;
  const int nt = tile / KT;
  const int kt = tile % KT;
  const int n = nt * 16 + (lane & 15);
  const int kb = kt * 32 + ((lane & 16) ? 8 : 0);
  bf16_t* out = dst + ((size_t)tile * 32 + lane) * 16;
  const float* src = W + (size_t)n * K + kb;
#pragma unroll
  for (int i = 0; i < 16; ++i) out[i] = (bf16_t)src[(i & 7) + ((i & 8) << 1)];
}

// ---------------------------------------------------------------------------
// WMMA GEMM:  C(M_ROWS x G3, bf16) = A(M_ROWS x K, bf16) * Bp^T + bias
// Block: 256 threads (8 waves, 2x4), block tile 64M x 256N, wave tile 32x64.
// Grid: (M_ROWS/64, G3/256).
// ---------------------------------------------------------------------------
__global__ __launch_bounds__(256) void gemm_wmma_kernel(
    const bf16_t* __restrict__ A, const bf16_t* __restrict__ Bp,
    const float* __restrict__ bias, bf16_t* __restrict__ Cout, int K) {
  const int KT = K >> 5;
  const int lane = threadIdx.x & 31;
  const int wv = threadIdx.x >> 5;
  const int wM = wv & 1, wN = wv >> 1;
  const int m_base = blockIdx.x * 64 + wM * 32;
  const int n_base = blockIdx.y * 256 + wN * 64;
  const int l15 = lane & 15;
  const int lh8 = (lane & 16) ? 8 : 0;

  v8f acc[2][4];
#pragma unroll
  for (int mf = 0; mf < 2; ++mf)
#pragma unroll
    for (int nf = 0; nf < 4; ++nf)
#pragma unroll
      for (int v = 0; v < 8; ++v) acc[mf][nf][v] = 0.0f;

  for (int kt = 0; kt < KT; ++kt) {
    const int kb = kt * 32 + lh8;
    v16bf a[2];
#pragma unroll
    for (int mf = 0; mf < 2; ++mf) {
      const int row = m_base + mf * 16 + l15;
      const v8bf* pa = (const v8bf*)(A + (size_t)row * K + kb);
      a[mf] = cat16(pa[0], pa[2]);  // k {0..7} and {16..23} of this chunk
    }
#pragma unroll
    for (int nf = 0; nf < 4; ++nf) {
      const int nt = (n_base >> 4) + nf;
      const v8bf* pb = (const v8bf*)(Bp + (((size_t)nt * KT + kt) * 32 + lane) * 16);
      const v16bf b = cat16(pb[0], pb[1]);
#pragma unroll
      for (int mf = 0; mf < 2; ++mf) acc[mf][nf] = wmma_bf16(a[mf], b, acc[mf][nf]);
    }
  }
  // epilogue: + bias, store bf16
#pragma unroll
  for (int nf = 0; nf < 4; ++nf) {
    const int n = n_base + nf * 16 + l15;
    const float bn = bias[n];
#pragma unroll
    for (int mf = 0; mf < 2; ++mf) {
      const int m0 = m_base + mf * 16 + lh8;
#pragma unroll
      for (int v = 0; v < 8; ++v)
        Cout[(size_t)(m0 + v) * G3 + n] = (bf16_t)(acc[mf][nf][v] + bn);
    }
  }
}

// ---------------------------------------------------------------------------
// Persistent GRU scan. grid.x = 2 (dir 0 fwd / 1 bwd), block = 1024 (32 waves).
// Wave w owns columns j in [16w,16w+16): gate-aligned N-tiles {w, 32+w, 64+w},
// so r/z/n accumulators stay in registers. h double-buffered bf16 in LDS.
// LAYER==0 -> bf16 output into y_bf ; LAYER==1 -> fp32 output into y_f32.
// ---------------------------------------------------------------------------
template <int LAYER>
__global__ __launch_bounds__(1024) void gru_scan_kernel(
    const bf16_t* __restrict__ giF, const bf16_t* __restrict__ giB,
    const bf16_t* __restrict__ whhPF, const bf16_t* __restrict__ whhPB,
    const float* __restrict__ bhhF, const float* __restrict__ bhhB,
    bf16_t* __restrict__ y_bf, float* __restrict__ y_f32,
    float* __restrict__ hid_out) {
  extern __shared__ bf16_t hsm[];  // 2 x [32][SP]
  const int dir = blockIdx.x;
  const bf16_t* gi = dir ? giB : giF;
  const bf16_t* whhP = dir ? whhPB : whhPF;
  const float* bhh = dir ? bhhB : bhhF;

  const int lane = threadIdx.x & 31;
  const int w = threadIdx.x >> 5;  // 0..31
  const int l15 = lane & 15;
  const int lh8 = (lane & 16) ? 8 : 0;
  const int j = w * 16 + l15;  // hidden column owned by this lane

  for (int i = threadIdx.x; i < 2 * 32 * SP; i += 1024) hsm[i] = (bf16_t)0.0f;
  __syncthreads();

  const float bhr = bhh[j], bhz = bhh[HID + j], bhn = bhh[2 * HID + j];
  const int KT = HID / 32;  // 16
  int cur = 0;

  for (int step = 0; step < T_STEPS; ++step) {
    const int t = dir ? (T_STEPS - 1 - step) : step;
    const bf16_t* hb = hsm + cur * (32 * SP);
    bf16_t* hn_buf = hsm + (cur ^ 1) * (32 * SP);

    v8f accr[2], accz[2], accn[2];
#pragma unroll
    for (int mf = 0; mf < 2; ++mf)
#pragma unroll
      for (int v = 0; v < 8; ++v) {
        accr[mf][v] = 0.0f; accz[mf][v] = 0.0f; accn[mf][v] = 0.0f;
      }

    for (int kt = 0; kt < KT; ++kt) {
      const int kb = kt * 32 + lh8;
      const v8bf* pa0 = (const v8bf*)(hb + (size_t)l15 * SP + kb);
      const v8bf* pa1 = (const v8bf*)(hb + (size_t)(16 + l15) * SP + kb);
      const v16bf a0 = cat16(pa0[0], pa0[2]);
      const v16bf a1 = cat16(pa1[0], pa1[2]);
#pragma unroll
      for (int g = 0; g < 3; ++g) {
        const int nt = g * 32 + w;
        const v8bf* pb = (const v8bf*)(whhP + (((size_t)nt * KT + kt) * 32 + lane) * 16);
        const v16bf b = cat16(pb[0], pb[1]);
        v8f* ag = (g == 0) ? accr : (g == 1) ? accz : accn;
        ag[0] = wmma_bf16(a0, b, ag[0]);
        ag[1] = wmma_bf16(a1, b, ag[1]);
      }
    }

    // gate epilogue: fully register-resident per lane
#pragma unroll
    for (int mf = 0; mf < 2; ++mf) {
#pragma unroll
      for (int v = 0; v < 8; ++v) {
        const int bt = mf * 16 + lh8 + v;  // batch row
        const size_t girow = ((size_t)t * BATCH + bt) * G3;
        const float gr = (float)gi[girow + j];
        const float gz = (float)gi[girow + HID + j];
        const float gn = (float)gi[girow + 2 * HID + j];
        const float r = sigmoidf_(gr + accr[mf][v] + bhr);
        const float z = sigmoidf_(gz + accz[mf][v] + bhz);
        const float nn = tanhf(gn + r * (accn[mf][v] + bhn));
        const float hp = (float)hb[(size_t)bt * SP + j];
        const float hnew = (1.0f - z) * nn + z * hp;
        hn_buf[(size_t)bt * SP + j] = (bf16_t)hnew;
        const size_t yi = ((size_t)t * BATCH + bt) * 1024 + (size_t)dir * HID + j;
        if (LAYER == 0) y_bf[yi] = (bf16_t)hnew;
        else            y_f32[yi] = hnew;
        if (step == T_STEPS - 1)
          hid_out[((size_t)(LAYER * 2 + dir)) * BATCH * HID + (size_t)bt * HID + j] = hnew;
      }
    }
    __syncthreads();
    cur ^= 1;
  }
}

// ---------------------------------------------------------------------------
extern "C" void kernel_launch(void* const* d_in, const int* in_sizes, int n_in,
                              void* d_out, int out_size, void* d_ws, size_t ws_size,
                              hipStream_t stream) {
  const float* X     = (const float*)d_in[0];
  const float* Wih0f = (const float*)d_in[1];  const float* Whh0f = (const float*)d_in[2];
  const float* bih0f = (const float*)d_in[3];  const float* bhh0f = (const float*)d_in[4];
  const float* Wih0b = (const float*)d_in[5];  const float* Whh0b = (const float*)d_in[6];
  const float* bih0b = (const float*)d_in[7];  const float* bhh0b = (const float*)d_in[8];
  const float* Wih1f = (const float*)d_in[9];  const float* Whh1f = (const float*)d_in[10];
  const float* bih1f = (const float*)d_in[11]; const float* bhh1f = (const float*)d_in[12];
  const float* Wih1b = (const float*)d_in[13]; const float* Whh1b = (const float*)d_in[14];
  const float* bih1b = (const float*)d_in[15]; const float* bhh1b = (const float*)d_in[16];

  char* ws = (char*)d_ws;
  size_t off = 0;
  auto wsa = [&](size_t bytes) -> char* {
    char* p = ws + off;
    off += (bytes + 255) & ~(size_t)255;
    return p;
  };
  bf16_t* Xbf  = (bf16_t*)wsa((size_t)M_ROWS * DIM * 2);    // 16 MB
  bf16_t* Y0   = (bf16_t*)wsa((size_t)M_ROWS * 1024 * 2);   // 32 MB (layer-0 output, bf16)
  bf16_t* GIF  = (bf16_t*)wsa((size_t)M_ROWS * G3 * 2);     // 48 MB (reused across layers)
  bf16_t* GIB  = (bf16_t*)wsa((size_t)M_ROWS * G3 * 2);
  bf16_t* WP0f = (bf16_t*)wsa((size_t)G3 * 512 * 2);
  bf16_t* WP0b = (bf16_t*)wsa((size_t)G3 * 512 * 2);
  bf16_t* WP1f = (bf16_t*)wsa((size_t)G3 * 1024 * 2);
  bf16_t* WP1b = (bf16_t*)wsa((size_t)G3 * 1024 * 2);
  bf16_t* WH0f = (bf16_t*)wsa((size_t)G3 * 512 * 2);
  bf16_t* WH0b = (bf16_t*)wsa((size_t)G3 * 512 * 2);
  bf16_t* WH1f = (bf16_t*)wsa((size_t)G3 * 512 * 2);
  bf16_t* WH1b = (bf16_t*)wsa((size_t)G3 * 512 * 2);

  // convert X to bf16
  f32_to_bf16_kernel<<<4096, 256, 0, stream>>>(X, Xbf, M_ROWS * DIM);

  // pack all weights into WMMA fragment order
  const int pk512 = ((G3 / 16) * (512 / 32) * 32 + 255) / 256;   // 192 blocks
  const int pk1024 = ((G3 / 16) * (1024 / 32) * 32 + 255) / 256; // 384 blocks
  pack_w_kernel<<<pk512, 256, 0, stream>>>(Wih0f, WP0f, 512);
  pack_w_kernel<<<pk512, 256, 0, stream>>>(Wih0b, WP0b, 512);
  pack_w_kernel<<<pk1024, 256, 0, stream>>>(Wih1f, WP1f, 1024);
  pack_w_kernel<<<pk1024, 256, 0, stream>>>(Wih1b, WP1b, 1024);
  pack_w_kernel<<<pk512, 256, 0, stream>>>(Whh0f, WH0f, 512);
  pack_w_kernel<<<pk512, 256, 0, stream>>>(Whh0b, WH0b, 512);
  pack_w_kernel<<<pk512, 256, 0, stream>>>(Whh1f, WH1f, 512);
  pack_w_kernel<<<pk512, 256, 0, stream>>>(Whh1b, WH1b, 512);

  const dim3 gg(M_ROWS / 64, G3 / 256);  // (256, 6)
  const size_t shmem = (size_t)2 * 32 * SP * sizeof(bf16_t);  // 66,560 B
  float* out = (float*)d_out;
  float* hid = out + (size_t)M_ROWS * 1024;

  // layer 0: input projections + scan
  gemm_wmma_kernel<<<gg, 256, 0, stream>>>(Xbf, WP0f, bih0f, GIF, 512);
  gemm_wmma_kernel<<<gg, 256, 0, stream>>>(Xbf, WP0b, bih0b, GIB, 512);
  gru_scan_kernel<0><<<2, 1024, shmem, stream>>>(GIF, GIB, WH0f, WH0b, bhh0f, bhh0b,
                                                 Y0, (float*)nullptr, hid);

  // layer 1: input projections (K=1024 from bf16 layer-0 output) + scan
  gemm_wmma_kernel<<<gg, 256, 0, stream>>>(Y0, WP1f, bih1f, GIF, 1024);
  gemm_wmma_kernel<<<gg, 256, 0, stream>>>(Y0, WP1b, bih1b, GIB, 1024);
  gru_scan_kernel<1><<<2, 1024, shmem, stream>>>(GIF, GIB, WH1f, WH1b, bhh1f, bhh1b,
                                                 (bf16_t*)nullptr, out, hid);
}
